// Encoder_15771119911584
// MI455X (gfx1250) — compile-verified, matrix operand order checked by reading, not run on hardware
//
#include <hip/hip_runtime.h>
#include <hip/hip_bf16.h>

typedef __attribute__((ext_vector_type(16))) _Float16 v16h;
typedef __attribute__((ext_vector_type(8)))  float    v8f;
typedef __attribute__((ext_vector_type(4)))  int      v4i;

#define EPS_BN 1e-5f

// --- optional CDNA5 async global->LDS path (guarded so compile never breaks) ---
#if defined(__has_builtin)
#if __has_builtin(__builtin_amdgcn_global_load_async_to_lds_b128) && \
    __has_builtin(__builtin_amdgcn_s_wait_asynccnt)
#define USE_ASYNC_LDS 1
#endif
#endif

// ---------------------------------------------------------------------------
// helpers
// ---------------------------------------------------------------------------
__device__ __forceinline__ int reflect_idx(int i, int n) {
    if (i < 0) i = -i;
    if (i >= n) i = 2 * n - 2 - i;
    return i;
}

// ---------------------------------------------------------------------------
// weight cast: f32 OIHW (flattened [O][K]) -> f16 [O][Kpad], zero-padded K
// ---------------------------------------------------------------------------
__global__ void cast_w_kernel(const float* __restrict__ w, _Float16* __restrict__ out,
                              int O, int K, int Kpad) {
    size_t i = (size_t)blockIdx.x * blockDim.x + threadIdx.x;
    size_t tot = (size_t)O * Kpad;
    if (i >= tot) return;
    int o = (int)(i / Kpad);
    int k = (int)(i % Kpad);
    out[i] = (k < K) ? (_Float16)w[(size_t)o * K + k] : (_Float16)0.0f;
}

__global__ void cast_act_kernel(const float* __restrict__ x, _Float16* __restrict__ y, size_t n) {
    size_t i = (size_t)blockIdx.x * blockDim.x + threadIdx.x;
    if (i < n) y[i] = (_Float16)x[i];
}

// ---------------------------------------------------------------------------
// implicit-GEMM conv with WMMA (wave32, v_wmma_f32_16x16x32_f16)
//   block = 256 threads (8 waves); block tile M=64 (out ch) x N=128 (pixels)
//   wave (wm,wn): wm -> 16 M-rows, wn -> 64 N-cols = four 16x16 WMMA tiles
//   double-buffered LDS; A tile staged with async global->LDS when available
//   mode 0: zero pad/stride s; mode 1: reflect pad; mode 2: lhs_dilation=2
// ---------------------------------------------------------------------------
__global__ void __launch_bounds__(256)
conv_wmma_kernel(const _Float16* __restrict__ A16, const _Float16* __restrict__ X16,
                 const float* __restrict__ bias, float* __restrict__ Y,
                 int Nb, int inC, int H, int W, int outC, int OH, int OW,
                 int KH, int KW, int stride, int pad, int mode, int Kpad) {
    __shared__ _Float16 As[2][64][32];    // [buf][m][k]   4 KB each
    __shared__ _Float16 Bs[2][128][32];   // [buf][n][k]   8 KB each

    const int tid  = threadIdx.x;
    const int lane = tid & 31;
    const int wave = tid >> 5;   // 0..7
    const int wm   = wave >> 1;  // 0..3 -> M sub-row
    const int wn   = wave & 1;   // 0..1 -> N half

    const int mBase = blockIdx.y * 64;
    const int nBase = blockIdx.x * 128;
    const int OHW   = OH * OW;
    const int Ntot  = Nb * OHW;
    const int K     = inC * KH * KW;
    const int HW    = H * W;
    const int KHW   = KH * KW;

    v8f acc[4] = {};

    // ---- cooperative-load coordinates ----
    const int aRow = tid >> 2;          // 0..63
    const int aCol = (tid & 3) * 8;     // 0,8,16,24
    const int ocA  = mBase + aRow;
    const _Float16* aSrcRow = A16 + (size_t)ocA * Kpad;

    // B: 128 rows x 32 k = 512 8-half chunks; 2 chunks per thread
    int bRowA[2], bColA[2], bNv[2], oyv[2], oxv[2];
    bool nOkA[2];
#pragma unroll
    for (int q = 0; q < 2; ++q) {
        const int chunk = tid + q * 256;
        bRowA[q] = chunk >> 2;
        bColA[q] = (chunk & 3) * 8;
        const int nIdx = nBase + bRowA[q];
        nOkA[q] = (nIdx < Ntot);
        bNv[q] = 0; oyv[q] = 0; oxv[q] = 0;
        if (nOkA[q]) {
            bNv[q] = nIdx / OHW;
            const int s = nIdx - bNv[q] * OHW;
            oyv[q] = s / OW;
            oxv[q] = s - oyv[q] * OW;
        }
    }

    auto load_tile = [&](int k0, int buf) {
        // ---- A tile (weights, contiguous) ----
        if (ocA < outC) {
#if defined(USE_ASYNC_LDS)
            __builtin_amdgcn_global_load_async_to_lds_b128(
                (v4i*)(aSrcRow + k0 + aCol),
                (v4i*)(void*)&As[buf][aRow][aCol], 0, 0);
#else
            const _Float16* src = aSrcRow + k0 + aCol;
            __builtin_prefetch(src + 32, 0, 3);
#pragma unroll
            for (int j = 0; j < 8; ++j) As[buf][aRow][aCol + j] = src[j];
#endif
        } else {
#pragma unroll
            for (int j = 0; j < 8; ++j) As[buf][aRow][aCol + j] = (_Float16)0.0f;
        }
        // ---- B tile (im2col gather) ----
#pragma unroll
        for (int q = 0; q < 2; ++q) {
            const int br = bRowA[q], bc = bColA[q];
#pragma unroll
            for (int j = 0; j < 8; ++j) {
                _Float16 v = (_Float16)0.0f;
                const int k = k0 + bc + j;
                if (nOkA[q] && k < K) {
                    const int ic = k / KHW;
                    const int r  = k - ic * KHW;
                    const int ky = r / KW;
                    const int kx = r - ky * KW;
                    int iy, ix;
                    bool ok = true;
                    if (mode == 2) {
                        const int dy = oyv[q] + ky - 1, dx = oxv[q] + kx - 1;
                        ok = ((dy | dx) >= 0) && !((dy | dx) & 1);
                        iy = dy >> 1; ix = dx >> 1;
                        ok = ok && (iy < H) && (ix < W);
                    } else {
                        iy = oyv[q] * stride + ky - pad;
                        ix = oxv[q] * stride + kx - pad;
                        if (mode == 1) { iy = reflect_idx(iy, H); ix = reflect_idx(ix, W); }
                        else ok = (iy >= 0) && (iy < H) && (ix >= 0) && (ix < W);
                    }
                    if (ok) v = X16[((size_t)bNv[q] * inC + ic) * HW + (size_t)iy * W + ix];
                }
                Bs[buf][br][bc + j] = v;
            }
        }
    };

    const int l  = lane & 15;
    const int kb = (lane >> 4) * 8;   // lanes 16..31 -> K halves +8/+24

    auto compute = [&](int buf) {
        v16h a{};
        const int m = wm * 16 + l;
#pragma unroll
        for (int j = 0; j < 8; ++j) {
            a[j]     = As[buf][m][kb + j];
            a[8 + j] = As[buf][m][16 + kb + j];
        }
#pragma unroll
        for (int t = 0; t < 4; ++t) {
            const int n = wn * 64 + t * 16 + l;
            v16h b{};
#pragma unroll
            for (int j = 0; j < 8; ++j) {
                b[j]     = Bs[buf][n][kb + j];
                b[8 + j] = Bs[buf][n][16 + kb + j];
            }
            acc[t] = __builtin_amdgcn_wmma_f32_16x16x32_f16(
                false, a, false, b, (short)0, acc[t], false, false);
        }
    };

    // ---- double-buffered main loop ----
    int buf = 0;
    load_tile(0, buf);
#if defined(USE_ASYNC_LDS)
    __builtin_amdgcn_s_wait_asynccnt(0);
#endif
    __syncthreads();
    for (int k0 = 0; k0 < Kpad; k0 += 32) {
        const int nxt = k0 + 32;
        if (nxt < Kpad) load_tile(nxt, buf ^ 1);
        compute(buf);
#if defined(USE_ASYNC_LDS)
        __builtin_amdgcn_s_wait_asynccnt(0);
#endif
        __syncthreads();
        buf ^= 1;
    }

    // ---- epilogue: bias add + store (C/D layout: VGPR r -> M=r(+8), lane -> N) ----
    const int hi = lane >> 4;
#pragma unroll
    for (int t = 0; t < 4; ++t) {
        const int n = nBase + wn * 64 + t * 16 + l;
        if (n >= Ntot) continue;
        const int bb = n / OHW;
        const int s  = n - bb * OHW;
        const size_t rowBase = ((size_t)bb * outC) * OHW + s;
#pragma unroll
        for (int r = 0; r < 8; ++r) {
            const int oc = mBase + wm * 16 + r + hi * 8;
            if (oc >= outC) continue;
            Y[rowBase + (size_t)oc * OHW] = acc[t][r] + bias[oc];
        }
    }
}

// ---------------------------------------------------------------------------
// BatchNorm batch statistics, two-stage: partial (atomic) + finalize
// ---------------------------------------------------------------------------
__global__ void __launch_bounds__(256)
bn_partial_kernel(const float* __restrict__ X, float* __restrict__ sumB,
                  float* __restrict__ sqB, int Nb, int C, int HW) {
    const int c = blockIdx.x;
    const int slice = blockIdx.y;
    const int S = gridDim.y;
    const int tot = Nb * HW;
    float s = 0.0f, s2 = 0.0f;
    for (int i = slice * 256 + threadIdx.x; i < tot; i += S * 256) {
        const int n  = i / HW;
        const int sp = i - n * HW;
        const float v = X[((size_t)n * C + c) * HW + sp];
        s += v;
        s2 += v * v;
    }
    __shared__ float sh[256], sh2[256];
    sh[threadIdx.x] = s;
    sh2[threadIdx.x] = s2;
    __syncthreads();
    for (int o = 128; o > 0; o >>= 1) {
        if (threadIdx.x < o) {
            sh[threadIdx.x]  += sh[threadIdx.x + o];
            sh2[threadIdx.x] += sh2[threadIdx.x + o];
        }
        __syncthreads();
    }
    if (threadIdx.x == 0) {
        atomicAdd(&sumB[c], sh[0]);
        atomicAdd(&sqB[c], sh2[0]);
    }
}

__global__ void bn_finalize_kernel(const float* __restrict__ sumB, const float* __restrict__ sqB,
                                   float* __restrict__ mean, float* __restrict__ var,
                                   int C, float invTot) {
    int c = blockIdx.x * blockDim.x + threadIdx.x;
    if (c >= C) return;
    const float m = sumB[c] * invTot;
    mean[c] = m;
    var[c]  = sqB[c] * invTot - m * m;
}

// fused scale/shift + ReLU; writes f32 feat (optional) and f16 for next conv
__global__ void bn_apply_kernel(const float* __restrict__ X, const float* __restrict__ mean,
                                const float* __restrict__ var, const float* __restrict__ g,
                                const float* __restrict__ b, float* __restrict__ Yf,
                                _Float16* __restrict__ Yh, int C, int HW, size_t total) {
    size_t i = (size_t)blockIdx.x * blockDim.x + threadIdx.x;
    if (i >= total) return;
    const int c = (int)((i / (size_t)HW) % (size_t)C);
    float v = (X[i] - mean[c]) * rsqrtf(var[c] + EPS_BN) * g[c] + b[c];
    v = fmaxf(v, 0.0f);
    if (Yf) Yf[i] = v;
    Yh[i] = (_Float16)v;
}

__global__ void tanh_kernel(const float* __restrict__ x, float* __restrict__ y, size_t n) {
    size_t i = (size_t)blockIdx.x * blockDim.x + threadIdx.x;
    if (i < n) y[i] = tanhf(x[i]);
}

__global__ void zero_kernel(float* __restrict__ p, int n) {
    int i = blockIdx.x * blockDim.x + threadIdx.x;
    if (i < n) p[i] = 0.0f;
}

// ---------------------------------------------------------------------------
// instance-wise mean: LDS-aggregated segment sums (ds_add_f32), then global
// atomics once per (block,label); scatter kernel divides by counts.
// Each 256-pixel block lies within one batch image (HW % 256 == 0).
// ---------------------------------------------------------------------------
__global__ void __launch_bounds__(256)
inst_sum_kernel(const float* __restrict__ out3, const int* __restrict__ inst,
                float* __restrict__ sums, float* __restrict__ cnt, int HW) {
    __shared__ float ssum[32][3];
    __shared__ float scnt[32];
    const int tid = threadIdx.x;
    if (tid < 32) {
        ssum[tid][0] = 0.0f; ssum[tid][1] = 0.0f; ssum[tid][2] = 0.0f;
        scnt[tid] = 0.0f;
    }
    __syncthreads();
    const size_t i = (size_t)blockIdx.x * 256 + tid;
    const int b = (int)(i / HW);
    const int s = (int)(i - (size_t)b * HW);
    const int lbl = inst[i] & 31;
#pragma unroll
    for (int c = 0; c < 3; ++c)
        atomicAdd(&ssum[lbl][c], out3[((size_t)b * 3 + c) * HW + s]);
    atomicAdd(&scnt[lbl], 1.0f);
    __syncthreads();
    if (tid < 32 && scnt[tid] > 0.0f) {
        const int seg = b * 32 + tid;
#pragma unroll
        for (int c = 0; c < 3; ++c) atomicAdd(&sums[seg * 3 + c], ssum[tid][c]);
        atomicAdd(&cnt[seg], scnt[tid]);
    }
}

__global__ void inst_scatter_kernel(const float* __restrict__ sums, const float* __restrict__ cnt,
                                    const int* __restrict__ inst, float* __restrict__ om,
                                    int HW) {
    size_t i = (size_t)blockIdx.x * blockDim.x + threadIdx.x;
    if (i >= (size_t)4 * HW) return;
    const int b = (int)(i / HW);
    const int s = (int)(i - (size_t)b * HW);
    const int seg = b * 32 + (inst[i] & 31);
    const float cc = fmaxf(cnt[seg], 1.0f);
#pragma unroll
    for (int c = 0; c < 3; ++c)
        om[((size_t)b * 3 + c) * HW + s] = sums[seg * 3 + c] / cc;
}

// ---------------------------------------------------------------------------
// host orchestration
// ---------------------------------------------------------------------------
struct ConvLayer {
    int wi, bi, gi, bei;
    int inC, H, W, outC, OH, OW;
    int KH, KW, stride, pad, mode;
    int K, Kpad;
    long long woff;
    long long featOff;
};

extern "C" void kernel_launch(void* const* d_in, const int* in_sizes, int n_in,
                              void* d_out, int out_size, void* d_ws, size_t ws_size,
                              hipStream_t stream) {
    (void)in_sizes; (void)n_in; (void)out_size; (void)ws_size;

    const float* x    = (const float*)d_in[0];
    const int*   inst = (const int*)d_in[1];
    float*       dout = (float*)d_out;

    // workspace layout (bytes)
    char* ws = (char*)d_ws;
    float*    bufA  = (float*)(ws);                           // 33.55M f32 conv out
    float*    bufB  = (float*)(ws + 134217728ll);             // 33.55M f32 (tanh out)
    _Float16* act16 = (_Float16*)(ws + 268435456ll);          // 33.55M f16 activations
    _Float16* w16   = (_Float16*)(ws + 335544320ll);          // f16 weight pool
    float*    stats = (float*)(ws + 343932928ll);             // mean|var|sum|sq (512 each)
    float*    isum  = (float*)(ws + 343941120ll);             // sums[384] | cnt[128]
    float*    meanP = stats;
    float*    varP  = stats + 512;
    float*    sumP  = stats + 1024;
    float*    sqP   = stats + 1536;
    float*    cntP  = isum + 384;

    static const ConvLayer L[9] = {
        //  wi bi  gi  bei inC   H    W  outC   OH   OW KH KW st pad mode   K  Kpad   woff      featOff
        {  2,  3,  4,  5,   3, 512, 512,  32, 512, 512, 7, 7, 1, 3, 1,  147,  160,       0, -1       },
        {  6, 10, 14, 18,  32, 512, 512,  64, 256, 256, 3, 3, 2, 1, 0,  288,  288,    5120, -1       },
        {  7, 11, 15, 19,  64, 256, 256, 128, 128, 128, 3, 3, 2, 1, 0,  576,  576,   23552, -1       },
        {  8, 12, 16, 20, 128, 128, 128, 256,  64,  64, 3, 3, 2, 1, 0, 1152, 1152,   97280, -1       },
        {  9, 13, 17, 21, 256,  64,  64, 512,  32,  32, 3, 3, 2, 1, 0, 2304, 2304,  392192,  3145728 },
        { 22, 26, 30, 34, 512,  32,  32, 256,  64,  64, 3, 3, 1, 1, 2, 4608, 4608, 1571840,  5242880 },
        { 23, 27, 31, 35, 256,  64,  64, 128, 128, 128, 3, 3, 1, 1, 2, 2304, 2304, 2751488,  9437184 },
        { 24, 28, 32, 36, 128, 128, 128,  64, 256, 256, 3, 3, 1, 1, 2, 1152, 1152, 3046400, 17825792 },
        { 25, 29, 33, 37,  64, 256, 256,  32, 512, 512, 3, 3, 1, 1, 2,  576,  576, 3120128, 34603008 },
    };
    const int LF_K = 1568, LF_Kpad = 1568;
    const long long LF_woff = 3138560ll;

    // 1) cast input activations to f16
    {
        size_t n = (size_t)4 * 3 * 512 * 512;
        cast_act_kernel<<<(unsigned)((n + 255) / 256), 256, 0, stream>>>(x, act16, n);
    }
    // 2) cast all weights to f16 [O][Kpad]
    for (int i = 0; i < 9; ++i) {
        size_t tot = (size_t)L[i].outC * L[i].Kpad;
        cast_w_kernel<<<(unsigned)((tot + 255) / 256), 256, 0, stream>>>(
            (const float*)d_in[L[i].wi], w16 + L[i].woff, L[i].outC, L[i].K, L[i].Kpad);
    }
    {
        size_t tot = (size_t)3 * LF_Kpad;
        cast_w_kernel<<<(unsigned)((tot + 255) / 256), 256, 0, stream>>>(
            (const float*)d_in[38], w16 + LF_woff, 3, LF_K, LF_Kpad);
    }

    // 3) conv + BN + ReLU pipeline
    for (int i = 0; i < 9; ++i) {
        const ConvLayer& c = L[i];
        const int Ntot = 4 * c.OH * c.OW;
        dim3 grid((Ntot + 127) / 128, (c.outC + 63) / 64);
        conv_wmma_kernel<<<grid, 256, 0, stream>>>(
            w16 + c.woff, act16, (const float*)d_in[c.bi], bufA,
            4, c.inC, c.H, c.W, c.outC, c.OH, c.OW,
            c.KH, c.KW, c.stride, c.pad, c.mode, c.Kpad);

        zero_kernel<<<4, 256, 0, stream>>>(sumP, 1024);  // zero sum+sq
        dim3 sgrid(c.outC, 32);
        bn_partial_kernel<<<sgrid, 256, 0, stream>>>(bufA, sumP, sqP, 4, c.outC, c.OH * c.OW);
        bn_finalize_kernel<<<(c.outC + 255) / 256, 256, 0, stream>>>(
            sumP, sqP, meanP, varP, c.outC, 1.0f / (float)(4 * c.OH * c.OW));

        size_t total = (size_t)4 * c.outC * c.OH * c.OW;
        float* featPtr = (c.featOff >= 0) ? (dout + c.featOff) : nullptr;
        bn_apply_kernel<<<(unsigned)((total + 255) / 256), 256, 0, stream>>>(
            bufA, meanP, varP, (const float*)d_in[c.gi], (const float*)d_in[c.bei],
            featPtr, act16, c.outC, c.OH * c.OW, total);
    }

    // 4) final reflect-pad 7x7 conv -> tanh
    {
        const int Ntot = 4 * 512 * 512;
        dim3 grid((Ntot + 127) / 128, 1);
        conv_wmma_kernel<<<grid, 256, 0, stream>>>(
            w16 + LF_woff, act16, (const float*)d_in[39], bufA,
            4, 32, 512, 512, 3, 512, 512, 7, 7, 1, 3, 1, LF_Kpad);
        size_t n = (size_t)4 * 3 * 512 * 512;
        tanh_kernel<<<(unsigned)((n + 255) / 256), 256, 0, stream>>>(bufA, bufB, n);
    }

    // 5) instance means
    {
        const int HW = 512 * 512;
        zero_kernel<<<2, 256, 0, stream>>>(isum, 512);
        size_t n = (size_t)4 * HW;
        inst_sum_kernel<<<(unsigned)(n / 256), 256, 0, stream>>>(bufB, inst, isum, cntP, HW);
        inst_scatter_kernel<<<(unsigned)((n + 255) / 256), 256, 0, stream>>>(
            isum, cntP, inst, dout, HW);
    }
}
